// SDPAWrapper_2216203124921
// MI455X (gfx1250) — matrix operand order repaired and checked by reading, not compiled, not run on hardware
//
#include <hip/hip_runtime.h>
#include <hip/hip_bf16.h>

typedef __attribute__((ext_vector_type(16))) _Float16 v16h;
typedef __attribute__((ext_vector_type(8)))  float    v8f;
typedef __attribute__((ext_vector_type(2)))  __fp16   fp16x2;

union Frag16 { v16h v; unsigned int u[8]; uint4 q[2]; };
union FragF  { v8f  v; float f[8]; };

#define KT_STRIDE 136   // halves per K-tile row (128 + 8 pad): rows 272B -> every fragment 16B aligned
#define VT_STRIDE 40    // halves per Vt row    (32 + 8 pad):  rows 80B  -> every fragment 16B aligned

__device__ __forceinline__ unsigned int pack2h(float a, float b) {
  union { fp16x2 h; unsigned int u; } p;
  p.h = __builtin_amdgcn_cvt_pkrtz(a, b);   // v_cvt_pk_rtz_f16_f32
  return p.u;
}

__global__ __launch_bounds__(128)
void fa_blockcausal_wmma(const float* __restrict__ Q, const float* __restrict__ K,
                         const float* __restrict__ V, const int* __restrict__ cu,
                         int ncu, int S, float* __restrict__ Out)
{
  __shared__ unsigned short kt[32 * KT_STRIDE];          // K tile, [key][dim] f16
  __shared__ unsigned short vt[128 * VT_STRIDE];         // V tile, [dim][key] f16 (transposed)

  const int tid  = threadIdx.x;
  const int lane = tid & 31;
  const int nn   = lane & 15;          // query index within wave tile (S^T col / O col)
  const int hi   = lane >> 4;          // half-wave select

  const int h   = blockIdx.y;
  const int qb  = blockIdx.x * 64;     // block's first query row
  const int q0w = qb + (tid >> 5) * 16;  // wave's first query row

  // Softmax scale folded into Q at f16-conversion time (base-2 domain).
  const float sc = 1.4426950408889634f / sqrtf(128.0f);  // log2(e)/sqrt(D)

  // Per-lane scalar doc info for query (q0w + nn)
  const int qi = q0w + nn;
  int st_lane = 0;
  for (int t = 0; t < ncu; ++t) { int c = cu[t]; if (c <= qi) st_lane = c; }
  const unsigned span = (unsigned)(qi - st_lane);   // allowed iff (unsigned)(j - st) <= span

  // Wave-uniform doc starts of first/last row (for early-out & mask-skip)
  int st_first = 0, st_last = 0, bst = 0;
  for (int t = 0; t < ncu; ++t) {
    int c = cu[t];
    if (c <= q0w)      st_first = c;
    if (c <= q0w + 15) st_last  = c;
    if (c <= qb)       bst      = c;
  }
  const int kb0 = bst & ~31;
  const int kb1 = (qb + 63) & ~31;

  // Hoisted per-lane LDS fragment base pointers (loop-invariant; DS 16-bit
  // immediate offsets reach every fragment from these two bases).
  const unsigned short* ktb = &kt[nn * KT_STRIDE + 8 * hi];
  const unsigned short* vtb = &vt[nn * VT_STRIDE + 8 * hi];

  // Persistent Q B-fragments (32x16 f16, pre-scaled by sc), 4 dim-chunks covering D=128.
  // B layout: lane -> col n = lane&15; VGPR v holds k pair k0 = 2v + 8*hi + 8*(v>>2)
  Frag16 bq[4];
  {
    const float* qrow = Q + ((size_t)(h * S + qi)) * 128;
#pragma unroll
    for (int c = 0; c < 4; ++c)
#pragma unroll
      for (int v = 0; v < 8; ++v) {
        int kk = 32 * c + 2 * v + 8 * hi + 8 * (v >> 2);
        bq[c].u[v] = pack2h(qrow[kk] * sc, qrow[kk + 1] * sc);
      }
  }

  // Ones B-fragment for WMMA row-sum (all entries 1.0h, layout-independent)
  Frag16 ones;
#pragma unroll
  for (int v = 0; v < 8; ++v) ones.u[v] = 0x3C003C00u;

  FragF o[8];                          // O accum, C-layout: comp r ↔ query row r+8*hi
#pragma unroll
  for (int j = 0; j < 8; ++j)
#pragma unroll
    for (int r = 0; r < 8; ++r) o[j].f[r] = 0.0f;
  float lr[8];                         // row sums, O-layout
#pragma unroll
  for (int r = 0; r < 8; ++r) lr[r] = 0.0f;
  float mr = -1e30f;                   // running max, per-lane scalar (query nn)

  for (int kb = kb0; kb <= kb1; kb += 32) {
    // ---- Cooperative staging (block-wide): K row-major, V transposed, f32 -> f16 ----
    for (int e = tid; e < 32 * 32; e += 128) {
      int key = e >> 5, grp = e & 31;
      const float4 k4 = ((const float4*)(K + ((size_t)(h * S + kb + key)) * 128))[grp];
      unsigned long long w = (unsigned long long)pack2h(k4.x, k4.y)
                           | ((unsigned long long)pack2h(k4.z, k4.w) << 32);
      *(unsigned long long*)&kt[key * KT_STRIDE + grp * 4] = w;
    }
    for (int e = tid; e < 128 * 16; e += 128) {
      int d = e & 127, key0 = (e >> 7) * 2;
      const float* vp = V + ((size_t)(h * S + kb + key0)) * 128 + d;
      *(unsigned int*)&vt[d * VT_STRIDE + key0] = pack2h(vp[0], vp[128]);
    }
    if (kb + 32 <= kb1) {
      int prow = tid >> 2, pseg = (tid & 3) << 5;
      size_t poff = ((size_t)(h * S + kb + 32 + prow)) * 128 + pseg;
      __builtin_prefetch(K + poff, 0, 1);
      __builtin_prefetch(V + poff, 0, 1);
    }
    __syncthreads();

    // Per-wave early-out: tile entirely above causal row or before our document
    const bool active = (kb <= q0w + 15) && (kb + 31 >= st_first);
    if (active) {
      // ---- S^T = K * (sc*Q)^T : comp r ↔ key kb + 16t + r + 8*hi, lane ↔ query nn ----
      // A fragment (tile t, chunk c): two contiguous ds_load_b128 at
      //   ktb + t*16*KT_STRIDE + c*32 (+16)
      FragF st0, st1;
#pragma unroll
      for (int r = 0; r < 8; ++r) { st0.f[r] = 0.0f; st1.f[r] = 0.0f; }

      Frag16 a0[2], a1[2];
      a0[0].q[0] = *(const uint4*)(ktb);
      a0[0].q[1] = *(const uint4*)(ktb + 16);
      a1[0].q[0] = *(const uint4*)(ktb + 16 * KT_STRIDE);
      a1[0].q[1] = *(const uint4*)(ktb + 16 * KT_STRIDE + 16);
#pragma unroll
      for (int c = 0; c < 4; ++c) {
        if (c + 1 < 4) {
          int off = (c + 1) * 32;
          a0[(c + 1) & 1].q[0] = *(const uint4*)(ktb + off);
          a0[(c + 1) & 1].q[1] = *(const uint4*)(ktb + off + 16);
          a1[(c + 1) & 1].q[0] = *(const uint4*)(ktb + 16 * KT_STRIDE + off);
          a1[(c + 1) & 1].q[1] = *(const uint4*)(ktb + 16 * KT_STRIDE + off + 16);
        }
        st0.v = __builtin_amdgcn_wmma_f32_16x16x32_f16(false, a0[c & 1].v, false, bq[c].v,
                                                       (short)0, st0.v, false, false);
        st1.v = __builtin_amdgcn_wmma_f32_16x16x32_f16(false, a1[c & 1].v, false, bq[c].v,
                                                       (short)0, st1.v, false, false);
      }

      // ---- Masking (boundary tiles only): key j = kb + 16t + r + 8hi vs lane's doc span ----
      const bool boundary = (kb + 31 > q0w) || (kb < st_last);
      if (boundary) {
        int base = kb + 8 * hi - st_lane;
#pragma unroll
        for (int r = 0; r < 8; ++r) {
          if ((unsigned)(base + r)      > span) st0.f[r] = -INFINITY;
          if ((unsigned)(base + 16 + r) > span) st1.f[r] = -INFINITY;
        }
      }

      // ---- Tile max: in-lane over 16 comps + one xor-16 shuffle ----
      float m = fmaxf(st0.f[0], st1.f[0]);
#pragma unroll
      for (int r = 1; r < 8; ++r) m = fmaxf(m, fmaxf(st0.f[r], st1.f[r]));
      m = fmaxf(m, __shfl_xor(m, 16, 32));

      // ---- Conditional rescale: only when some lane's running max increased ----
      if (__any(m > mr)) {
        float mnew = fmaxf(mr, m);                       // >= -1e30, always finite
        float alpha = __builtin_amdgcn_exp2f(mr - mnew); // raw v_exp_f32
        mr = mnew;
        float ar[8];
#pragma unroll
        for (int r = 0; r < 8; ++r) ar[r] = __shfl(alpha, r + 8 * hi, 32);  // to O-layout
#pragma unroll
        for (int r = 0; r < 8; ++r) lr[r] *= ar[r];
#pragma unroll
        for (int j = 0; j < 8; ++j)
#pragma unroll
          for (int r = 0; r < 8; ++r) o[j].f[r] *= ar[r];
      }

      // ---- P = exp2(x - m): in-register, then pack straight into the A-fragment ----
      float p0[8], p1[8];
#pragma unroll
      for (int r = 0; r < 8; ++r) {
        p0[r] = __builtin_amdgcn_exp2f(st0.f[r] - mr);   // -inf -> 0 for masked keys
        p1[r] = __builtin_amdgcn_exp2f(st1.f[r] - mr);
      }
      Frag16 ap;   // A pair v: tile v>>2, comps 2(v&3), 2(v&3)+1  (pure in-lane!)
#pragma unroll
      for (int v = 0; v < 8; ++v) {
        int c0 = (v & 3) * 2;
        ap.u[v] = (v < 4) ? pack2h(p0[c0], p0[c0 + 1]) : pack2h(p1[c0], p1[c0 + 1]);
      }

      // Row-sum of P via WMMA: lands in exactly the O-layout slots of lr[]
      {
        FragF lsum;
#pragma unroll
        for (int r = 0; r < 8; ++r) lsum.f[r] = 0.0f;
        lsum.v = __builtin_amdgcn_wmma_f32_16x16x32_f16(false, ap.v, false, ones.v,
                                                        (short)0, lsum.v, false, false);
#pragma unroll
        for (int r = 0; r < 8; ++r) lr[r] += lsum.f[r];
      }

      // ---- O += P * V : 8 dim-tiles, K=32 keys; double-buffered b128 fragment loads ----
      Frag16 bv[2];
      bv[0].q[0] = *(const uint4*)(vtb);
      bv[0].q[1] = *(const uint4*)(vtb + 16);
#pragma unroll
      for (int j = 0; j < 8; ++j) {
        if (j + 1 < 8) {
          int off = (j + 1) * 16 * VT_STRIDE;
          bv[(j + 1) & 1].q[0] = *(const uint4*)(vtb + off);
          bv[(j + 1) & 1].q[1] = *(const uint4*)(vtb + off + 16);
        }
        o[j].v = __builtin_amdgcn_wmma_f32_16x16x32_f16(false, ap.v, false, bv[j & 1].v,
                                                        (short)0, o[j].v, false, false);
      }
    }
    __syncthreads();   // protect kt/vt before next staging round
  }

  // ---- Epilogue: normalize and store fp32 (O-layout: row r+8hi, col 16j+nn) ----
#pragma unroll
  for (int r = 0; r < 8; ++r) {
    float inv = 1.0f / lr[r];
    float* orow = Out + ((size_t)(h * S + q0w + r + 8 * hi)) * 128;
#pragma unroll
    for (int j = 0; j < 8; ++j) orow[16 * j + nn] = o[j].f[r] * inv;
  }
}

extern "C" void kernel_launch(void* const* d_in, const int* in_sizes, int n_in,
                              void* d_out, int out_size, void* d_ws, size_t ws_size,
                              hipStream_t stream) {
  const float* q  = (const float*)d_in[0];
  const float* k  = (const float*)d_in[1];
  const float* v  = (const float*)d_in[2];
  const int*   cu = (const int*)d_in[3];
  const int ncu = in_sizes[3];
  const int H = 16, D = 128;
  const int S = in_sizes[0] / (H * D);     // B=1
  float* out = (float*)d_out;

  dim3 grid(S / 64, H);
  dim3 block(128);
  fa_blockcausal_wmma<<<grid, block, 0, stream>>>(q, k, v, cu, ncu, S, out);
}
